// DEKTNet_31671088840783
// MI455X (gfx1250) — matrix operand order repaired and checked by reading, not compiled, not run on hardware
//
#include <hip/hip_runtime.h>
#include <math.h>

#define B_   32
#define T_   200
#define DK   128
#define NQ   100
#define NQP  112   // n_q padded to 7*16 for WMMA M tiles

typedef __attribute__((ext_vector_type(16))) __bf16         v16bf;
typedef __attribute__((ext_vector_type(8)))  float          v8f;
typedef __attribute__((ext_vector_type(8)))  unsigned short ushort8;

union BF16x16 { v16bf v; ushort8 h[2]; unsigned short u[16]; };

__device__ __forceinline__ float rcpf(float x){ return __builtin_amdgcn_rcpf(x); }
__device__ __forceinline__ float sigf(float x){ return rcpf(1.0f + __expf(-x)); }
__device__ __forceinline__ float tanh_fast(float x){ return 2.0f*sigf(2.0f*x) - 1.0f; }
__device__ __forceinline__ unsigned short f2bf(float f){
  unsigned u = __float_as_uint(f);
  u += 0x7FFFu + ((u>>16)&1u);          // round-to-nearest-even
  return (unsigned short)(u>>16);
}
__device__ __forceinline__ int mapcls(float x){
  int i = (int)(x*5000.0f);
  return i < 0 ? 0 : (i > 5000 ? 5000 : i);
}

// packed bf16 weight pair matvec segment: 64 k-pairs (=128 inputs)
// layout: wt[(pairBase+kp)*256 + 2*j] holds bf16{w[j][2kp], w[j][2kp+1]}
__device__ __forceinline__ float mv_seg(const unsigned short* wt, int pairBase,
                                        const float* x, int j, float acc){
  #pragma unroll 4
  for (int kp=0; kp<64; kp++){
    const unsigned uu = *(const unsigned*)(wt + (pairBase+kp)*256 + 2*j);
    acc += x[2*kp]   * __uint_as_float(uu << 16);
    acc += x[2*kp+1] * __uint_as_float(uu & 0xffff0000u);
  }
  return acc;
}
__device__ __forceinline__ float mv_seg_scalar(const unsigned short* wt, int pairBase,
                                               float xs, int j, float acc){
  #pragma unroll 4
  for (int kp=0; kp<64; kp++){
    const unsigned uu = *(const unsigned*)(wt + (pairBase+kp)*256 + 2*j);
    acc += xs * (__uint_as_float(uu << 16) + __uint_as_float(uu & 0xffff0000u));
  }
  return acc;
}

struct P {
  const int   *e_data,*s_data,*at_data,*it_data,*qd_data,*sd_data,*tp_data,*att_data;
  const float *a_data,*bor_data,*conc_data,*conf_data,*fru_data;
  const float *q_matrix,*h0,*affect_h0;
  const float *at_e,*it_e,*e_e,*s_e,*att_e,*fru_e,*conf_e,*conc_e,*bor_e,*sd_e,*qd_e,*tp_e;
  const float *w1,*b1,*w2,*b2,*w3,*b3,*w4,*b4,*w5,*b5,*wa,*ba,*we,*be,*watt,*batt,*wemo,*bemo;
  const float *wbor,*bbor,*wconc,*bconc,*wconf,*bconf,*wfru,*bfru,*wsig,*bsig,*wtan,*btan;
  float *emo_seq,*attblk,*learn_seq;                 // ws: sequence projections
  const unsigned short *t2,*t3,*t4b,*ta,*tsig,*ttan,*te,*t5; // ws: packed bf16 W^T
  float *out;
};

// ---------------------------------------------------------------------------
// Pack kernel: W[j][stride] cols [off, off+Kout) -> bf16 pairs [Kout/2][128][2]
// ---------------------------------------------------------------------------
__global__ void pack_wT(const float* __restrict__ w, unsigned short* __restrict__ dst,
                        int stride, int off, int Kout){
  const int kp = blockIdx.x;           // 0 .. Kout/2-1
  const int j  = threadIdx.x;          // 0 .. 127
  dst[kp*256 + 2*j]     = f2bf(w[j*stride + off + 2*kp]);
  dst[kp*256 + 2*j + 1] = f2bf(w[j*stride + off + 2*kp + 1]);
  (void)Kout;
}

// ---------------------------------------------------------------------------
// Kernel 1: sequence projections as WMMA GEMMs, 64 (b,t) rows per block.
//   out[64x128] = sigma-free linear: Xb(bf16 gathered)[64xK] @ W^T + bias
// ---------------------------------------------------------------------------
template <int K>
__device__ __forceinline__ void gemm64(const unsigned short* Xb, const float* w,
                                       const float* bias, float* outp, int bt0,
                                       int ln, int half, int dcol){
  v8f acc[4];
  const float bv = bias[dcol];
  #pragma unroll
  for (int m=0;m<4;m++){
    #pragma unroll
    for (int g=0;g<8;g++) acc[m][g] = bv;
  }
  for (int kk=0; kk<K/32; kk++){
    BF16x16 Bt;                                    // B[k][n] = w[n][k]
    const float* wp = &w[dcol*K + kk*32 + half*16];
    #pragma unroll
    for (int i=0;i<16;i++) Bt.u[i] = f2bf(wp[i]);
    #pragma unroll
    for (int m=0;m<4;m++){
      BF16x16 A;
      const int arow = m*16 + ln;
      const int kb = kk*32 + half*16;
      A.h[0] = *(const ushort8*)&Xb[arow*K + kb];
      A.h[1] = *(const ushort8*)&Xb[arow*K + kb + 8];
      acc[m] = __builtin_amdgcn_wmma_f32_16x16x32_bf16(
                 false, A.v, false, Bt.v, (short)0, acc[m], false, false);
    }
  }
  #pragma unroll
  for (int m=0;m<4;m++)
    #pragma unroll
    for (int g=0;g<8;g++)
      outp[(bt0 + m*16 + g + 8*half)*DK + dcol] = acc[m][g];
}

__global__ void dekt_precompute(P p){
  extern __shared__ char dynp[];
  unsigned short* einB = (unsigned short*)dynp;            // [64][512]
  unsigned short* ainB = einB + 64*512;                    // [64][640]
  unsigned short* linB = ainB + 64*640;                    // [64][512]
  const int bt0 = blockIdx.x*64;
  const int tid = threadIdx.x;
  const int d   = tid & 127;
  const int rr  = tid >> 7;
  for (int r = rr; r < 64; r += 2){
    const int bt = bt0 + r;
    const int e  = p.e_data[bt],  s  = p.s_data[bt], at = p.at_data[bt];
    const int qd = p.qd_data[bt], sd = p.sd_data[bt], tp = p.tp_data[bt];
    const float a = p.a_data[bt];
    const int fi = mapcls(p.fru_data[bt]),  ci = mapcls(p.conf_data[bt]);
    const int cc = mapcls(p.conc_data[bt]), bi = mapcls(p.bor_data[bt]);
    unsigned short* er = einB + r*512;
    er[d]      = f2bf(p.fru_e [fi*DK+d]);
    er[128+d]  = f2bf(p.conf_e[ci*DK+d]);
    er[256+d]  = f2bf(p.conc_e[cc*DK+d]);
    er[384+d]  = f2bf(p.bor_e [bi*DK+d]);
    unsigned short* ar = ainB + r*640;
    const float ee = p.e_e[e*DK+d], se = p.s_e[s*DK+d];
    ar[d]      = f2bf(ee);
    ar[128+d]  = f2bf(se);
    ar[256+d]  = f2bf(p.qd_e[qd*DK+d]);
    ar[384+d]  = f2bf(p.sd_e[sd*DK+d]);
    ar[512+d]  = f2bf(p.tp_e[tp*DK+d]);
    unsigned short* lr = linB + r*512;
    lr[d]      = f2bf(p.at_e[at*DK+d]);
    lr[128+d]  = f2bf(se);
    lr[256+d]  = f2bf(a);
    lr[384+d]  = f2bf(ee);
  }
  __syncthreads();
  const int lane = tid & 31, wave = tid >> 5;
  const int half = lane >> 4, ln = lane & 15;
  const int dcol = wave*16 + ln;
  gemm64<512>(einB, p.wemo, p.bemo, p.emo_seq,   bt0, ln, half, dcol);
  gemm64<640>(ainB, p.watt, p.batt, p.attblk,    bt0, ln, half, dcol);
  gemm64<512>(linB, p.w1,   p.b1,   p.learn_seq, bt0, ln, half, dcol);
}

// ---------------------------------------------------------------------------
// Kernel 2: persistent scan — one 256-thread (8-wave) workgroup per batch row.
// ---------------------------------------------------------------------------
__global__ void dekt_scan(P p){
  extern __shared__ char dyn[];
  float*          hf  = (float*)dyn;                        // 112*128 f32
  unsigned short* hb0 = (unsigned short*)(dyn + NQP*DK*4);  // bf16 buf A
  unsigned short* hb1 = hb0 + NQP*DK;                       // bf16 buf B

  __shared__ float qe[NQP], qen[NQP];
  __shared__ float emo[DK], blkA[DK], blkB[DK], learn[DK];
  __shared__ float atv[DK], itv[DK], aedv[DK];
  __shared__ float LG[DK], base_[DK], htil[DK], ah[DK];
  __shared__ float aff[DK], FLG[DK], xv[DK], res[DK];
  __shared__ float uhv[DK], bxv[DK];
  __shared__ float fb[4][DK];
  __shared__ float red[DK];
  __shared__ float soft4[4], xh[4], sc[4];

  const int b    = blockIdx.x;
  const int tid  = threadIdx.x;
  const int lane = tid & 31, wave = tid >> 5;
  const int half = lane >> 4, ln = lane & 15;
  const int dcol = wave*16 + ln;

  // Constant B-matrix tiles: W4h = l4.w[:,0:128], B[k][n] = w4[n][k] (bf16)
  BF16x16 Breg[4];
  #pragma unroll
  for (int kk=0;kk<4;kk++)
    #pragma unroll
    for (int i=0;i<16;i++)
      Breg[kk].u[i] = f2bf(p.w4[dcol*384 + kk*32 + half*16 + i]);

  for (int idx = tid; idx < NQP*DK; idx += 256){
    const int q = idx >> 7;
    const float v = (q < NQ) ? p.h0[idx] : 0.0f;
    hf[idx] = v; hb0[idx] = f2bf(v);
  }
  if (tid < DK){
    ah[tid] = p.affect_h0[b*DK + tid];
    const int e0 = p.e_data[b*T_];
    float acc = 0.0f;
    for (int q=0;q<NQ;q++) acc += p.q_matrix[e0*NQ+q] * p.h0[q*DK + tid];
    htil[tid] = acc;
  }
  if (tid < 5) p.out[tid*(B_*T_) + b*T_] = 0.0f;
  __syncthreads();

  int parity = 0;
  for (int t=0; t<T_-1; t++){
    unsigned short* hc = parity ? hb1 : hb0;
    unsigned short* hn = parity ? hb0 : hb1;
    const int bt = b*T_ + t;

    // ---- P0: gathers / streamed projections ----
    if (tid < DK){
      const int d = tid;
      emo[d]   = p.emo_seq[bt*DK+d];
      blkA[d]  = p.attblk[bt*DK+d];
      blkB[d]  = p.attblk[(bt+1)*DK+d];
      learn[d] = p.learn_seq[bt*DK+d];
      atv[d]   = p.at_e [p.at_data[bt]*DK+d];
      itv[d]   = p.it_e [p.it_data[bt]*DK+d];
      aedv[d]  = p.att_e[p.att_data[bt]*DK+d];
      if (d < NQP){
        const int ei = p.e_data[bt];
        qe[d] = (d < NQ) ? p.q_matrix[ei*NQ+d] : 0.0f;
      }
      if (t+2 < T_) __builtin_prefetch(&p.emo_seq[(bt+1)*DK+d], 0, 1);
    } else {
      const int d = tid - 128;
      if (d < NQP){
        const int e1 = p.e_data[bt+1];
        qen[d] = (d < NQ) ? p.q_matrix[e1*NQ+d] : 0.0f;
      }
    }
    if (tid == 0) sc[0] = p.a_data[bt];
    __syncthreads();

    // ---- P1: 4-way softmax over rel = [a, aed, at, it] against emo ----
    if (tid < 4){
      float acc = 0.0f;
      if (tid == 0){ const float a = sc[0]; for (int d=0;d<DK;d++) acc += emo[d]*a; }
      else {
        const float* rv = (tid==1) ? aedv : (tid==2) ? atv : itv;
        for (int d=0;d<DK;d++) acc += emo[d]*rv[d];
      }
      red[tid] = acc;
    }
    __syncthreads();
    if (tid == 0){
      const float m  = fmaxf(fmaxf(red[0],red[1]), fmaxf(red[2],red[3]));
      const float e0=__expf(red[0]-m), e1=__expf(red[1]-m),
                  e2=__expf(red[2]-m), e3=__expf(red[3]-m);
      const float rs = rcpf(e0+e1+e2+e3);
      soft4[0]=e0*rs; soft4[1]=e1*rs; soft4[2]=e2*rs; soft4[3]=e3*rs;
    }
    __syncthreads();
    if (tid < DK)
      res[tid] = soft4[0]*sc[0] + soft4[1]*aedv[tid] + soft4[2]*atv[tid] + soft4[3]*itv[tid];
    __syncthreads();

    // ---- P2: LG (l2,l3) and aff (la) — packed bf16 W^T, coalesced ----
    if (tid < DK){
      const int j = tid;
      float a2 = p.b2[j], a3 = p.b3[j], aa = p.ba[j];
      a2 = mv_seg(p.t2, 0, learn, j, a2);
      a2 = mv_seg(p.t2, 64, htil, j, a2);
      a3 = mv_seg(p.t3, 0, learn, j, a3);
      a3 = mv_seg(p.t3, 64, htil, j, a3);
      a3 = mv_seg(p.t3, 128, emo, j, a3);
      aa = mv_seg(p.ta, 0, emo, j, aa);
      aa = mv_seg(p.ta, 64, blkA, j, aa);
      aa = mv_seg(p.ta, 128, res, j, aa);
      aa = mv_seg_scalar(p.ta, 192, sc[0], j, aa);
      LG[j]  = sigf(a3) * sigf(2.0f*a2);      // sigmoid(a3)*(tanh(a2)+1)/2
      aff[j] = sigf(aa);
    }
    __syncthreads();

    // ---- P3: base (l4 LG/it part) and FLG (lsiga*ltana on cat4) ----
    if (tid < DK){
      const int j = tid;
      float ab = p.b4[j], as = p.bsig[j], at2 = p.btan[j];
      ab  = mv_seg(p.t4b, 0, LG, j, ab);
      ab  = mv_seg(p.t4b, 64, itv, j, ab);
      as  = mv_seg(p.tsig, 0, aff, j, as);
      as  = mv_seg(p.tsig, 64, res, j, as);
      as  = mv_seg_scalar(p.tsig, 128, sc[0], j, as);
      as  = mv_seg(p.tsig, 192, ah, j, as);
      at2 = mv_seg(p.ttan, 0, aff, j, at2);
      at2 = mv_seg(p.ttan, 64, res, j, at2);
      at2 = mv_seg_scalar(p.ttan, 128, sc[0], j, at2);
      at2 = mv_seg(p.ttan, 192, ah, j, at2);
      base_[j] = ab;
      FLG[j]   = sigf(as)*tanh_fast(at2);
    }
    __syncthreads();

    // ---- P4: w1 = softmax_d(LG*FLG); ah update ----
    if (tid < DK) red[tid] = LG[tid]*FLG[tid];
    __syncthreads();
    for (int s=64; s>0; s>>=1){ if (tid < s) red[tid] = fmaxf(red[tid], red[tid+s]); __syncthreads(); }
    if (tid == 0) sc[1] = red[0];
    __syncthreads();
    if (tid < DK) red[tid] = __expf(LG[tid]*FLG[tid] - sc[1]);
    __syncthreads();
    for (int s=64; s>0; s>>=1){ if (tid < s) red[tid] += red[tid+s]; __syncthreads(); }
    if (tid == 0) sc[2] = rcpf(red[0]);
    __syncthreads();
    if (tid < DK){
      const float w1 = __expf(LG[tid]*FLG[tid] - sc[1]) * sc[2];
      ah[tid] = (1.0f - w1)*FLG[tid] + w1*ah[tid];
    }
    __syncthreads();

    // ---- P5: x_ = sigmoid(le([ah, blkB])); affect heads; feedback gathers ----
    if (tid < DK){
      const int j = tid;
      float ae = p.be[j];
      ae = mv_seg(p.te, 0, ah, j, ae);
      ae = mv_seg(p.te, 64, blkB, j, ae);
      xv[j] = sigf(ae);
    }
    __syncthreads();
    if (tid < 4){
      const float* w  = (tid==0)? p.wbor : (tid==1)? p.wconc : (tid==2)? p.wconf : p.wfru;
      const float* bb = (tid==0)? p.bbor : (tid==1)? p.bconc : (tid==2)? p.bconf : p.bfru;
      float acc = bb[0];
      for (int i=0;i<32;i++) acc += xv[tid*32+i]*w[i];
      xh[tid] = sigf(acc);
    }
    __syncthreads();
    if (tid < DK){
      const int d = tid;
      fb[0][d] = p.fru_e [mapcls(xh[3])*DK+d];
      fb[1][d] = p.conf_e[mapcls(xh[2])*DK+d];
      fb[2][d] = p.conc_e[mapcls(xh[1])*DK+d];
      fb[3][d] = p.bor_e [mapcls(xh[0])*DK+d];
    }

    // ---- P6: WMMA GEMM  gmf = sigmoid(h_pre @ W4h^T + base);  h update ----
    {
      const float bval = base_[dcol];
      const float lgd  = LG[dcol];
      for (int m=0; m<7; m++){
        const int arow = m*16 + ln;
        BF16x16 A[4];
        #pragma unroll
        for (int kk=0; kk<4; kk++){
          const int kb = kk*32 + half*16;
          A[kk].h[0] = *(const ushort8*)&hc[arow*DK + kb];
          A[kk].h[1] = *(const ushort8*)&hc[arow*DK + kb + 8];
        }
        v8f c;
        #pragma unroll
        for (int g=0; g<8; g++) c[g] = bval;
        #pragma unroll
        for (int kk=0; kk<4; kk++)
          c = __builtin_amdgcn_wmma_f32_16x16x32_bf16(
                false, A[kk].v, false, Breg[kk].v, (short)0, c, false, false);
        #pragma unroll
        for (int g=0; g<8; g++){
          const int q = m*16 + g + half*8;
          const float gmf  = sigf(c[g]);
          const float hnew = qe[q]*lgd + gmf*hf[q*DK + dcol];
          hf[q*DK + dcol] = hnew;
          hn[q*DK + dcol] = f2bf(hnew);
        }
      }
    }
    __syncthreads();

    // ---- P7: h_tilde = q_e_nxt . h ;  stage uh / blkB*x_ ----
    if (tid < DK){
      float acc = 0.0f;
      for (int q=0;q<NQ;q++) acc += qen[q]*hf[q*DK + tid];
      htil[tid] = acc;
      uhv[tid] = acc*xv[tid];
      bxv[tid] = blkB[tid]*xv[tid];
    }
    __syncthreads();

    // ---- P8: y = mean(sigmoid(l5([fru,conf,conc,bor, uh, blkB*x_]))) ----
    if (tid < DK){
      const int j = tid;
      float a5 = p.b5[j];
      a5 = mv_seg(p.t5, 0,   fb[0], j, a5);
      a5 = mv_seg(p.t5, 64,  fb[1], j, a5);
      a5 = mv_seg(p.t5, 128, fb[2], j, a5);
      a5 = mv_seg(p.t5, 192, fb[3], j, a5);
      a5 = mv_seg(p.t5, 256, uhv,  j, a5);
      a5 = mv_seg(p.t5, 320, bxv,  j, a5);
      red[j] = sigf(a5);
    }
    __syncthreads();
    for (int s=64; s>0; s>>=1){ if (tid < s) red[tid] += red[tid+s]; __syncthreads(); }
    if (tid == 0) p.out[b*T_ + t + 1] = red[0] * (1.0f/(float)DK);
    if (tid < 4) p.out[(tid+1)*(B_*T_) + b*T_ + t + 1] = xh[tid];
    parity ^= 1;
    __syncthreads();
  }
}

// ---------------------------------------------------------------------------
extern "C" void kernel_launch(void* const* d_in, const int* in_sizes, int n_in,
                              void* d_out, int out_size, void* d_ws, size_t ws_size,
                              hipStream_t stream){
  (void)in_sizes; (void)n_in; (void)out_size; (void)ws_size;
  P p;
  p.e_data   =(const int*)  d_in[0];  p.s_data  =(const int*)  d_in[1];
  p.at_data  =(const int*)  d_in[2];  p.a_data  =(const float*)d_in[3];
  p.it_data  =(const int*)  d_in[4];  p.bor_data=(const float*)d_in[5];
  p.conc_data=(const float*)d_in[6];  p.conf_data=(const float*)d_in[7];
  p.fru_data =(const float*)d_in[8];  p.qd_data =(const int*)  d_in[9];
  p.sd_data  =(const int*)  d_in[10]; p.tp_data =(const int*)  d_in[11];
  /* 12: stu_data, 13: pre_data — unused by the forward pass */
  p.att_data =(const int*)  d_in[14];
  p.q_matrix =(const float*)d_in[15]; p.h0      =(const float*)d_in[16];
  p.affect_h0=(const float*)d_in[17];
  p.at_e =(const float*)d_in[18]; p.it_e =(const float*)d_in[19];
  p.e_e  =(const float*)d_in[20]; p.s_e  =(const float*)d_in[21];
  p.att_e=(const float*)d_in[22]; p.fru_e=(const float*)d_in[23];
  p.conf_e=(const float*)d_in[24]; p.conc_e=(const float*)d_in[25];
  p.bor_e=(const float*)d_in[26]; p.sd_e =(const float*)d_in[27];
  p.qd_e =(const float*)d_in[28]; p.tp_e =(const float*)d_in[29];
  p.w1=(const float*)d_in[30]; p.b1=(const float*)d_in[31];
  p.w2=(const float*)d_in[32]; p.b2=(const float*)d_in[33];
  p.w3=(const float*)d_in[34]; p.b3=(const float*)d_in[35];
  p.w4=(const float*)d_in[36]; p.b4=(const float*)d_in[37];
  p.w5=(const float*)d_in[38]; p.b5=(const float*)d_in[39];
  p.wa=(const float*)d_in[40]; p.ba=(const float*)d_in[41];
  p.we=(const float*)d_in[42]; p.be=(const float*)d_in[43];
  p.watt=(const float*)d_in[44]; p.batt=(const float*)d_in[45];
  p.wemo=(const float*)d_in[46]; p.bemo=(const float*)d_in[47];
  p.wbor=(const float*)d_in[48];  p.bbor=(const float*)d_in[49];
  p.wconc=(const float*)d_in[50]; p.bconc=(const float*)d_in[51];
  p.wconf=(const float*)d_in[52]; p.bconf=(const float*)d_in[53];
  p.wfru=(const float*)d_in[54];  p.bfru=(const float*)d_in[55];
  p.wsig=(const float*)d_in[56];  p.bsig=(const float*)d_in[57];
  p.wtan=(const float*)d_in[58];  p.btan=(const float*)d_in[59];

  float* wsf = (float*)d_ws;
  p.emo_seq   = wsf;
  p.attblk    = wsf +     B_*T_*DK;
  p.learn_seq = wsf + 2 * B_*T_*DK;
  unsigned short* wsu = (unsigned short*)(wsf + 3 * B_*T_*DK);
  unsigned short* u = wsu;
  unsigned short *t2=u;   u += 256*DK;
  unsigned short *t3=u;   u += 384*DK;
  unsigned short *t4b=u;  u += 256*DK;
  unsigned short *ta=u;   u += 512*DK;
  unsigned short *tsig=u; u += 512*DK;
  unsigned short *ttan=u; u += 512*DK;
  unsigned short *te=u;   u += 256*DK;
  unsigned short *t5=u;   u += 768*DK;
  p.t2=t2; p.t3=t3; p.t4b=t4b; p.ta=ta; p.tsig=tsig; p.ttan=ttan; p.te=te; p.t5=t5;
  p.out = (float*)d_out;

  // one-time packs of scan weights into bf16 W^T pairs
  pack_wT<<<128, 128, 0, stream>>>(p.w2,   t2,   256, 0,   256);
  pack_wT<<<192, 128, 0, stream>>>(p.w3,   t3,   384, 0,   384);
  pack_wT<<<128, 128, 0, stream>>>(p.w4,   t4b,  384, 128, 256);
  pack_wT<<<256, 128, 0, stream>>>(p.wa,   ta,   512, 0,   512);
  pack_wT<<<256, 128, 0, stream>>>(p.wsig, tsig, 512, 0,   512);
  pack_wT<<<256, 128, 0, stream>>>(p.wtan, ttan, 512, 0,   512);
  pack_wT<<<128, 128, 0, stream>>>(p.we,   te,   256, 0,   256);
  pack_wT<<<384, 128, 0, stream>>>(p.w5,   t5,   768, 0,   768);

  // sequence projections via WMMA, 64 rows per block
  const size_t ldsPre = (size_t)64*(512+640+512)*2;        // 208 KB bf16 inputs
  dekt_precompute<<<(B_*T_)/64, 256, ldsPre, stream>>>(p);

  // persistent per-batch scan
  const size_t ldsScan = (size_t)NQP*DK*4 + 2u*NQP*DK*2;   // 112 KB
  dekt_scan<<<B_, 256, ldsScan, stream>>>(p);
}
